// CausalAttention_37976100831500
// MI455X (gfx1250) — matrix operand order repaired and checked by reading, not compiled, not run on hardware
//
#include <hip/hip_runtime.h>
#include <math.h>

typedef __attribute__((ext_vector_type(16))) _Float16 v16h;
typedef __attribute__((ext_vector_type(8)))  _Float16 v8h;
typedef __attribute__((ext_vector_type(8)))  float    v8f;
typedef __attribute__((ext_vector_type(4)))  float    v4f;

#define D      1024
#define NSEQ   2048
#define NBATCH 4
#define MTOT   (NBATCH * NSEQ)   // 8192 flattened rows
#define NEGBIG (-3.0e4f)
#define LDK    40                // padded k-stride of LDS B tile (f16 units)

#if defined(__has_builtin)
#if __has_builtin(__builtin_amdgcn_sched_barrier)
#define SCHED_FENCE() __builtin_amdgcn_sched_barrier(0)
#endif
#endif
#ifndef SCHED_FENCE
#define SCHED_FENCE()
#endif

// Block tile: 256 rows x 128 cols, 8 waves in 4(M) x 2(N).
// Wave tile: 32 rows x 64 cols = 2 A-fragments x 4 B-fragments = 8 WMMA / K-tile,
// fed by just 8 ds_load_b128 (each B fragment reused by both A fragments).
// LDS B tile (double buffered, column-major): Bs[buf][n][k], n in [0,128), k in [0,32).

// ---------------------------------------------------------------------------
// Kernel 1: QKV projection.  C[m,n] = sum_k x[m,k] * W[k,n], f32 in, f16 out.
// grid = (MTOT/256, D/128, 3) ; block = 256.
// ---------------------------------------------------------------------------
__global__ __launch_bounds__(256) void qkv_gemm_kernel(
    const float* __restrict__ x,
    const float* __restrict__ Wq,
    const float* __restrict__ Wk,
    const float* __restrict__ Wv,
    _Float16* __restrict__ Qb,
    _Float16* __restrict__ Kb,
    _Float16* __restrict__ Vb)
{
    __shared__ _Float16 Bs[2][128][LDK];

    const int tid = threadIdx.x;
    const int l   = tid & 31;
    const int w   = tid >> 5;
    const int wm  = w & 3;
    const int wn  = w >> 2;
    const int r0  = blockIdx.x * 256;
    const int c0  = blockIdx.y * 128;

    const float* Wm;
    _Float16*    Out;
    if (blockIdx.z == 0)      { Wm = Wq; Out = Qb; }
    else if (blockIdx.z == 1) { Wm = Wk; Out = Kb; }
    else                      { Wm = Wv; Out = Vb; }

    const int base0 = (l < 16) ? 0 : 8;
    const int kb2   = (l < 16) ? 0 : 8;
    const int mrow0 = r0 + wm * 32 + (l & 15);
    const int ncb   = wn * 64 + (l & 15);

    const int sn  = tid & 127;
    const int sk0 = (tid >> 7) * 8;

    v8f acc[8];
#pragma unroll
    for (int t = 0; t < 8; ++t) acc[t] = (v8f)0.0f;

    v8h p0, p1;
    {
        const float* wp = Wm + (size_t)sk0 * D + c0 + sn;
#pragma unroll
        for (int j = 0; j < 8; ++j) p0[j] = (_Float16)wp[(size_t)j * D];
#pragma unroll
        for (int j = 0; j < 8; ++j) p1[j] = (_Float16)wp[(size_t)(16 + j) * D];
    }

    int buf = 0;
    for (int kb = 0; kb < D; kb += 32) {
        *(v8h*)&Bs[buf][sn][sk0]      = p0;
        *(v8h*)&Bs[buf][sn][sk0 + 16] = p1;
        __syncthreads();

        if (kb + 32 < D) {   // prefetch next tile while computing this one
            const float* wp = Wm + (size_t)(kb + 32 + sk0) * D + c0 + sn;
#pragma unroll
            for (int j = 0; j < 8; ++j) p0[j] = (_Float16)wp[(size_t)j * D];
#pragma unroll
            for (int j = 0; j < 8; ++j) p1[j] = (_Float16)wp[(size_t)(16 + j) * D];
        }

        // two A fragments (rows mrow0, mrow0+16), ISA 16-bit A layout
        v16h a[2];
#pragma unroll
        for (int i = 0; i < 2; ++i) {
            const float* ap = x + (size_t)(mrow0 + 16 * i) * D + kb + base0;
            v4f f0 = *(const v4f*)(ap);
            v4f f1 = *(const v4f*)(ap + 4);
            v4f f2 = *(const v4f*)(ap + 16);
            v4f f3 = *(const v4f*)(ap + 20);
#pragma unroll
            for (int j = 0; j < 4; ++j) {
                a[i][j]      = (_Float16)f0[j];
                a[i][4 + j]  = (_Float16)f1[j];
                a[i][8 + j]  = (_Float16)f2[j];
                a[i][12 + j] = (_Float16)f3[j];
            }
        }

        // batch 4 B fragments, fence, then 8 back-to-back WMMAs
        v16h bf[4];
#pragma unroll
        for (int t = 0; t < 4; ++t) {
            const int ncol = ncb + t * 16;
            v8h lo = *(const v8h*)&Bs[buf][ncol][kb2];
            v8h hi = *(const v8h*)&Bs[buf][ncol][kb2 + 16];
#pragma unroll
            for (int j = 0; j < 8; ++j) { bf[t][j] = lo[j]; bf[t][8 + j] = hi[j]; }
        }
        SCHED_FENCE();
#pragma unroll
        for (int t = 0; t < 4; ++t) {
            acc[t]     = __builtin_amdgcn_wmma_f32_16x16x32_f16(
                false, a[0], false, bf[t], (short)0, acc[t],     false, false);
            acc[4 + t] = __builtin_amdgcn_wmma_f32_16x16x32_f16(
                false, a[1], false, bf[t], (short)0, acc[4 + t], false, false);
        }

        buf ^= 1;
    }

#pragma unroll
    for (int i = 0; i < 2; ++i) {
#pragma unroll
        for (int t = 0; t < 4; ++t) {
#pragma unroll
            for (int r = 0; r < 8; ++r) {
                const int m = r0 + wm * 32 + i * 16 + ((l < 16) ? r : 8 + r);
                const int n = c0 + wn * 64 + t * 16 + (l & 15);
                Out[(size_t)m * D + n] = (_Float16)acc[i * 4 + t][r];
            }
        }
    }
}

// ---------------------------------------------------------------------------
// Kernel 2: scores S = (Q @ K^T) * 1/32 with causal mask, f16 out.
// grid = (NSEQ/256, NSEQ/128, NBATCH); block = 256.
// ---------------------------------------------------------------------------
__global__ __launch_bounds__(256) void scores_gemm_kernel(
    const _Float16* __restrict__ Qb,
    const _Float16* __restrict__ Kb,
    _Float16* __restrict__ Sb)
{
    __shared__ _Float16 Bs[2][128][LDK];

    const int tid = threadIdx.x;
    const int l   = tid & 31;
    const int w   = tid >> 5;
    const int wm  = w & 3;
    const int wn  = w >> 2;
    const int r0  = blockIdx.x * 256;
    const int c0  = blockIdx.y * 128;
    const int b   = blockIdx.z;

    _Float16* Srow = Sb + (size_t)b * NSEQ * NSEQ;

    // fully-masked tile: fill and bail (uniform branch)
    if (c0 >= r0 + 256) {
        v8h neg;
#pragma unroll
        for (int j = 0; j < 8; ++j) neg[j] = (_Float16)NEGBIG;
        for (int e = tid; e < (256 * 128) / 8; e += 256) {
            const int m = (e * 8) / 128;
            const int n = (e * 8) % 128;
            *(v8h*)(Srow + (size_t)(r0 + m) * NSEQ + c0 + n) = neg;
        }
        return;
    }

    const _Float16* Qg = Qb + (size_t)b * NSEQ * D;
    const _Float16* Kg = Kb + (size_t)b * NSEQ * D;

    const int base0 = (l < 16) ? 0 : 8;
    const int kb2   = (l < 16) ? 0 : 8;
    const int mrow0 = r0 + wm * 32 + (l & 15);
    const int ncb   = wn * 64 + (l & 15);

    const int sn  = tid >> 2;         // 0..63
    const int sk0 = (tid & 3) * 8;    // 0,8,16,24

    v8f acc[8];
#pragma unroll
    for (int t = 0; t < 8; ++t) acc[t] = (v8f)0.0f;

    v8h p0 = *(const v8h*)(Kg + (size_t)(c0 + sn) * D + sk0);
    v8h p1 = *(const v8h*)(Kg + (size_t)(c0 + sn + 64) * D + sk0);

    int buf = 0;
    for (int kb = 0; kb < D; kb += 32) {
        *(v8h*)&Bs[buf][sn][sk0]      = p0;
        *(v8h*)&Bs[buf][sn + 64][sk0] = p1;
        __syncthreads();

        if (kb + 32 < D) {
            p0 = *(const v8h*)(Kg + (size_t)(c0 + sn) * D + kb + 32 + sk0);
            p1 = *(const v8h*)(Kg + (size_t)(c0 + sn + 64) * D + kb + 32 + sk0);
        }

        v16h a[2];
#pragma unroll
        for (int i = 0; i < 2; ++i) {
            const _Float16* ap = Qg + (size_t)(mrow0 + 16 * i) * D + kb + base0;
            v8h lo = *(const v8h*)(ap);
            v8h hi = *(const v8h*)(ap + 16);
#pragma unroll
            for (int j = 0; j < 8; ++j) { a[i][j] = lo[j]; a[i][8 + j] = hi[j]; }
        }

        v16h bf[4];
#pragma unroll
        for (int t = 0; t < 4; ++t) {
            const int ncol = ncb + t * 16;
            v8h blo = *(const v8h*)&Bs[buf][ncol][kb2];
            v8h bhi = *(const v8h*)&Bs[buf][ncol][kb2 + 16];
#pragma unroll
            for (int j = 0; j < 8; ++j) { bf[t][j] = blo[j]; bf[t][8 + j] = bhi[j]; }
        }
        SCHED_FENCE();
#pragma unroll
        for (int t = 0; t < 4; ++t) {
            acc[t]     = __builtin_amdgcn_wmma_f32_16x16x32_f16(
                false, a[0], false, bf[t], (short)0, acc[t],     false, false);
            acc[4 + t] = __builtin_amdgcn_wmma_f32_16x16x32_f16(
                false, a[1], false, bf[t], (short)0, acc[4 + t], false, false);
        }

        buf ^= 1;
    }

    const float scale = 0.03125f;   // 1/sqrt(1024)
#pragma unroll
    for (int i = 0; i < 2; ++i) {
#pragma unroll
        for (int t = 0; t < 4; ++t) {
#pragma unroll
            for (int r = 0; r < 8; ++r) {
                const int m = r0 + wm * 32 + i * 16 + ((l < 16) ? r : 8 + r);
                const int n = c0 + wn * 64 + t * 16 + (l & 15);
                const float v = (n > m) ? NEGBIG : acc[i * 4 + t][r] * scale;
                Srow[(size_t)m * NSEQ + n] = (_Float16)v;
            }
        }
    }
}

// ---------------------------------------------------------------------------
// Kernel 3: in-place row softmax over S (f16), 2048 entries per row.
// grid = (NSEQ, NBATCH); block = 256, 8 elements per thread in registers.
// ---------------------------------------------------------------------------
__global__ __launch_bounds__(256) void softmax_rows_kernel(_Float16* __restrict__ Sb)
{
    __shared__ float red[256];
    const int tid = threadIdx.x;
    _Float16* row = Sb + ((size_t)blockIdx.y * NSEQ + blockIdx.x) * NSEQ;

    v8h sv = *(const v8h*)(row + tid * 8);
    float s[8];
#pragma unroll
    for (int j = 0; j < 8; ++j) s[j] = (float)sv[j];

    float m = -INFINITY;
#pragma unroll
    for (int j = 0; j < 8; ++j) m = fmaxf(m, s[j]);
    red[tid] = m;
    __syncthreads();
    for (int off = 128; off > 0; off >>= 1) {
        if (tid < off) red[tid] = fmaxf(red[tid], red[tid + off]);
        __syncthreads();
    }
    const float rowmax = red[0];
    __syncthreads();

    float p[8], sum = 0.0f;
#pragma unroll
    for (int j = 0; j < 8; ++j) { p[j] = __expf(s[j] - rowmax); sum += p[j]; }
    red[tid] = sum;
    __syncthreads();
    for (int off = 128; off > 0; off >>= 1) {
        if (tid < off) red[tid] += red[tid + off];
        __syncthreads();
    }
    const float inv = 1.0f / red[0];

    v8h pv;
#pragma unroll
    for (int j = 0; j < 8; ++j) pv[j] = (_Float16)(p[j] * inv);
    *(v8h*)(row + tid * 8) = pv;
}

// ---------------------------------------------------------------------------
// Kernel 4: O = P @ V  (f16 x f16 -> f32 out), K loop trimmed by causality.
// grid = (NSEQ/256, D/128, NBATCH); block = 256.
// ---------------------------------------------------------------------------
__global__ __launch_bounds__(256) void pv_gemm_kernel(
    const _Float16* __restrict__ Pb,
    const _Float16* __restrict__ Vb,
    float* __restrict__ Ob)
{
    __shared__ _Float16 Bs[2][128][LDK];

    const int tid = threadIdx.x;
    const int l   = tid & 31;
    const int w   = tid >> 5;
    const int wm  = w & 3;
    const int wn  = w >> 2;
    const int r0  = blockIdx.x * 256;
    const int c0  = blockIdx.y * 128;
    const int b   = blockIdx.z;

    const _Float16* Pg = Pb + (size_t)b * NSEQ * NSEQ;
    const _Float16* Vg = Vb + (size_t)b * NSEQ * D;
    float*          Og = Ob + (size_t)b * NSEQ * D;

    const int base0 = (l < 16) ? 0 : 8;
    const int kb2   = (l < 16) ? 0 : 8;
    const int mrow0 = r0 + wm * 32 + (l & 15);
    const int ncb   = wn * 64 + (l & 15);
    const int kend  = r0 + 256;   // cols > row are exactly 0 in P

    const int sn  = tid & 127;
    const int sk0 = (tid >> 7) * 8;

    v8f acc[8];
#pragma unroll
    for (int t = 0; t < 8; ++t) acc[t] = (v8f)0.0f;

    v8h p0, p1;
    {
        const _Float16* vp = Vg + (size_t)sk0 * D + c0 + sn;
#pragma unroll
        for (int j = 0; j < 8; ++j) p0[j] = vp[(size_t)j * D];
#pragma unroll
        for (int j = 0; j < 8; ++j) p1[j] = vp[(size_t)(16 + j) * D];
    }

    int buf = 0;
    for (int kb = 0; kb < kend; kb += 32) {
        *(v8h*)&Bs[buf][sn][sk0]      = p0;
        *(v8h*)&Bs[buf][sn][sk0 + 16] = p1;
        __syncthreads();

        if (kb + 32 < kend) {
            const _Float16* vp = Vg + (size_t)(kb + 32 + sk0) * D + c0 + sn;
#pragma unroll
            for (int j = 0; j < 8; ++j) p0[j] = vp[(size_t)j * D];
#pragma unroll
            for (int j = 0; j < 8; ++j) p1[j] = vp[(size_t)(16 + j) * D];
        }

        v16h a[2];
#pragma unroll
        for (int i = 0; i < 2; ++i) {
            const _Float16* ap = Pg + (size_t)(mrow0 + 16 * i) * NSEQ + kb + base0;
            v8h lo = *(const v8h*)(ap);
            v8h hi = *(const v8h*)(ap + 16);
#pragma unroll
            for (int j = 0; j < 8; ++j) { a[i][j] = lo[j]; a[i][8 + j] = hi[j]; }
        }

        v16h bf[4];
#pragma unroll
        for (int t = 0; t < 4; ++t) {
            const int ncol = ncb + t * 16;
            v8h blo = *(const v8h*)&Bs[buf][ncol][kb2];
            v8h bhi = *(const v8h*)&Bs[buf][ncol][kb2 + 16];
#pragma unroll
            for (int j = 0; j < 8; ++j) { bf[t][j] = blo[j]; bf[t][8 + j] = bhi[j]; }
        }
        SCHED_FENCE();
#pragma unroll
        for (int t = 0; t < 4; ++t) {
            acc[t]     = __builtin_amdgcn_wmma_f32_16x16x32_f16(
                false, a[0], false, bf[t], (short)0, acc[t],     false, false);
            acc[4 + t] = __builtin_amdgcn_wmma_f32_16x16x32_f16(
                false, a[1], false, bf[t], (short)0, acc[4 + t], false, false);
        }

        buf ^= 1;
    }

#pragma unroll
    for (int i = 0; i < 2; ++i) {
#pragma unroll
        for (int t = 0; t < 4; ++t) {
#pragma unroll
            for (int r = 0; r < 8; ++r) {
                const int m = r0 + wm * 32 + i * 16 + ((l < 16) ? r : 8 + r);
                const int n = c0 + wn * 64 + t * 16 + (l & 15);
                Og[(size_t)m * D + n] = acc[i * 4 + t][r];
            }
        }
    }
}

// ---------------------------------------------------------------------------
extern "C" void kernel_launch(void* const* d_in, const int* in_sizes, int n_in,
                              void* d_out, int out_size, void* d_ws, size_t ws_size,
                              hipStream_t stream) {
    (void)in_sizes; (void)n_in; (void)out_size; (void)ws_size;

    const float* x  = (const float*)d_in[0];
    const float* Wq = (const float*)d_in[1];
    const float* Wk = (const float*)d_in[2];
    const float* Wv = (const float*)d_in[3];
    float* out = (float*)d_out;

    const size_t SLOT = (size_t)MTOT * D;         // 8192*1024 f16 elements
    _Float16* Qb = (_Float16*)d_ws;
    _Float16* Kb = Qb + SLOT;
    _Float16* Vb = Kb + SLOT;
    _Float16* Sb = Vb + SLOT;                     // 4*2048*2048 f16 (S, then P in place)

    dim3 blk(256, 1, 1);

    dim3 g1(MTOT / 256, D / 128, 3);
    qkv_gemm_kernel<<<g1, blk, 0, stream>>>(x, Wq, Wk, Wv, Qb, Kb, Vb);

    dim3 g2(NSEQ / 256, NSEQ / 128, NBATCH);
    scores_gemm_kernel<<<g2, blk, 0, stream>>>(Qb, Kb, Sb);

    dim3 g3(NSEQ, NBATCH, 1);
    softmax_rows_kernel<<<g3, blk, 0, stream>>>(Sb);

    dim3 g4(NSEQ / 256, D / 128, NBATCH);
    pv_gemm_kernel<<<g4, blk, 0, stream>>>(Sb, Vb, out);
}
